// JiTDenoisingTransformer_21887153341032
// MI455X (gfx1250) — compile-verified, matrix operand order checked by reading, not run on hardware
//
#include <hip/hip_runtime.h>

// ---------------------------------------------------------------------------
// CDNA5 (gfx1250) DiT forward: bf16 WMMA GEMMs + fused elementwise kernels.
// All matrix math goes through V_WMMA_F32_16X16X32_BF16 (wave32).
// GEMM K-loop is register-double-buffered: next tile's global loads are in
// flight while the current tile is consumed by WMMA from LDS.
// ---------------------------------------------------------------------------

typedef __attribute__((ext_vector_type(16))) __bf16 v16bf;
typedef __attribute__((ext_vector_type(8)))  float  v8f;

struct Frag { union { v16bf v; uint4 q[2]; }; };

__device__ __forceinline__ unsigned short f2bf(float f) {
    unsigned int u = __float_as_uint(f);
    unsigned int r = u + 0x7FFFu + ((u >> 16) & 1u);   // round-to-nearest-even
    return (unsigned short)(r >> 16);
}

// ---------------------------------------------------------------------------
// Generic batched GEMM: C[M,N] (f32) = A[M,K](bf16) * B[K,N](bf16) + bias
// b_nk == 1 -> B is stored [N,K] row-major (i.e. already transposed).
// Tile: BM=128 x BN=128 x BK=32. 256 threads = 8 wave32 waves,
// each wave computes a 32x64 block as 2x4 WMMA 16x16 tiles.
// ---------------------------------------------------------------------------
#define GBM 128
#define GBN 128
#define GBK 32
#define GPAD 8   // row pitch (32+8)*2 = 80B: 16B-aligned, bank-spreading

__global__ __launch_bounds__(256) void gemm_bf16_kernel(
    const unsigned short* __restrict__ A,
    const unsigned short* __restrict__ Bm,
    const float* __restrict__ bias,
    float* __restrict__ C,
    int M, int N, int K,
    int lda, int ldb, int ldc,
    long long aBatch, long long bBatch, long long cBatch,
    int b_nk)
{
    __shared__ __align__(16) unsigned short lA[GBM][GBK + GPAD];
    __shared__ __align__(16) unsigned short lB[GBN][GBK + GPAD];

    int z = blockIdx.z;
    A  += (long long)z * aBatch;
    Bm += (long long)z * bBatch;
    C  += (long long)z * cBatch;

    int m0  = blockIdx.y * GBM;
    int n0  = blockIdx.x * GBN;
    int tid = threadIdx.x;
    int lane = tid & 31;
    int wid  = tid >> 5;
    int wm = (wid & 3) * 32;   // 4 waves along M -> 128 rows
    int wn = (wid >> 2) * 64;  // 2 waves along N -> 128 cols

    v8f acc[2][4];
#pragma unroll
    for (int a = 0; a < 2; ++a)
#pragma unroll
        for (int b = 0; b < 4; ++b)
#pragma unroll
            for (int r = 0; r < 8; ++r) acc[a][b][r] = 0.f;

    bool fastA = (m0 + GBM <= M) && ((lda & 7) == 0);
    bool fastB = (n0 + GBN <= N) && (b_nk ? ((ldb & 7) == 0) : ((ldb & 3) == 0));

    int lr = lane & 15;
    int kh = (lane >> 4) * 8;
    int nk = (K + GBK - 1) / GBK;

    // fragment loads + 8 WMMAs for the staged tile
    auto compute = [&]() {
        Frag fa[2], fb[4];
#pragma unroll
        for (int mt = 0; mt < 2; ++mt) {
            int row = wm + mt * 16 + lr;
            fa[mt].q[0] = *(const uint4*)&lA[row][kh];
            fa[mt].q[1] = *(const uint4*)&lA[row][kh + 16];
        }
#pragma unroll
        for (int nt = 0; nt < 4; ++nt) {
            int col = wn + nt * 16 + lr;
            fb[nt].q[0] = *(const uint4*)&lB[col][kh];
            fb[nt].q[1] = *(const uint4*)&lB[col][kh + 16];
        }
#pragma unroll
        for (int mt = 0; mt < 2; ++mt)
#pragma unroll
            for (int nt = 0; nt < 4; ++nt)
                acc[mt][nt] = __builtin_amdgcn_wmma_f32_16x16x32_bf16(
                    false, fa[mt].v, false, fb[nt].v,
                    (short)0, acc[mt][nt], false, false);
    };

    if (fastA && fastB && ((K & (GBK - 1)) == 0)) {
        // ---- pipelined path: double-buffer tiles through registers -------
        int ar = tid >> 2, aseg = (tid & 3) * 8;   // 8-bf16 chunk coords
        const unsigned short* aP = A + (long long)(m0 + ar) * lda + aseg;
        const unsigned short* aQ = aP + (long long)64 * lda;
        int bkk = tid >> 5, bns = (tid & 31) * 4;  // transpose-stage coords
        const unsigned short* bP = Bm;
        const unsigned short* bQ = Bm;
        const unsigned short* bT = Bm;
        if (b_nk) {
            bP = Bm + (long long)(n0 + ar) * ldb + aseg;
            bQ = bP + (long long)64 * ldb;
        } else {
            bT = Bm + (long long)bkk * ldb + n0 + bns;
        }

        uint4 ra0, ra1, rb0, rb1;
        uint2 rc0, rc1, rc2, rc3;

        // prologue: load + commit tile 0
        ra0 = *(const uint4*)aP; ra1 = *(const uint4*)aQ;
        if (b_nk) { rb0 = *(const uint4*)bP; rb1 = *(const uint4*)bQ; }
        else {
            rc0 = *(const uint2*)(bT);
            rc1 = *(const uint2*)(bT + (long long)8  * ldb);
            rc2 = *(const uint2*)(bT + (long long)16 * ldb);
            rc3 = *(const uint2*)(bT + (long long)24 * ldb);
        }
        auto commit = [&]() {
            *(uint4*)&lA[ar][aseg]      = ra0;
            *(uint4*)&lA[ar + 64][aseg] = ra1;
            if (b_nk) {
                *(uint4*)&lB[ar][aseg]      = rb0;
                *(uint4*)&lB[ar + 64][aseg] = rb1;
            } else {
                uint2 d;
                d = rc0;
                lB[bns + 0][bkk] = (unsigned short)(d.x & 0xffffu);
                lB[bns + 1][bkk] = (unsigned short)(d.x >> 16);
                lB[bns + 2][bkk] = (unsigned short)(d.y & 0xffffu);
                lB[bns + 3][bkk] = (unsigned short)(d.y >> 16);
                d = rc1;
                lB[bns + 0][bkk + 8] = (unsigned short)(d.x & 0xffffu);
                lB[bns + 1][bkk + 8] = (unsigned short)(d.x >> 16);
                lB[bns + 2][bkk + 8] = (unsigned short)(d.y & 0xffffu);
                lB[bns + 3][bkk + 8] = (unsigned short)(d.y >> 16);
                d = rc2;
                lB[bns + 0][bkk + 16] = (unsigned short)(d.x & 0xffffu);
                lB[bns + 1][bkk + 16] = (unsigned short)(d.x >> 16);
                lB[bns + 2][bkk + 16] = (unsigned short)(d.y & 0xffffu);
                lB[bns + 3][bkk + 16] = (unsigned short)(d.y >> 16);
                d = rc3;
                lB[bns + 0][bkk + 24] = (unsigned short)(d.x & 0xffffu);
                lB[bns + 1][bkk + 24] = (unsigned short)(d.x >> 16);
                lB[bns + 2][bkk + 24] = (unsigned short)(d.y & 0xffffu);
                lB[bns + 3][bkk + 24] = (unsigned short)(d.y >> 16);
            }
        };
        commit();

        for (int kb = 0; kb < nk; ++kb) {
            __syncthreads();
            bool more = (kb + 1) < nk;
            long long ko = (long long)(kb + 1) * GBK;
            if (more) {
                // issue next tile's global loads before consuming this tile
                ra0 = *(const uint4*)(aP + ko);
                ra1 = *(const uint4*)(aQ + ko);
                if (b_nk) {
                    rb0 = *(const uint4*)(bP + ko);
                    rb1 = *(const uint4*)(bQ + ko);
                } else {
                    const unsigned short* bt = bT + ko * ldb;
                    rc0 = *(const uint2*)(bt);
                    rc1 = *(const uint2*)(bt + (long long)8  * ldb);
                    rc2 = *(const uint2*)(bt + (long long)16 * ldb);
                    rc3 = *(const uint2*)(bt + (long long)24 * ldb);
                }
            }
            if (kb + 2 < nk) {
                long long ko2 = (long long)(kb + 2) * GBK;
                __builtin_prefetch(aP + ko2, 0, 0);
                if (b_nk) __builtin_prefetch(bP + ko2, 0, 0);
                else      __builtin_prefetch(bT + ko2 * ldb, 0, 0);
            }
            compute();
            __syncthreads();
            if (more) commit();
        }
    } else {
        // ---- fallback path: per-iteration guarded staging -----------------
        for (int kb = 0; kb < nk; ++kb) {
            int k0 = kb * GBK;
            bool fk = (k0 + GBK <= K);
            __syncthreads();
            if (fastA && fk) {
#pragma unroll
                for (int it = 0; it < 2; ++it) {
                    int c = tid + it * 256;
                    int r = c >> 2, seg = (c & 3) * 8;
                    *(uint4*)&lA[r][seg] =
                        *(const uint4*)(A + (long long)(m0 + r) * lda + k0 + seg);
                }
            } else {
                for (int i = tid; i < GBM * GBK; i += 256) {
                    int r  = i >> 5, kk = i & 31;
                    int gm = m0 + r, gk = k0 + kk;
                    lA[r][kk] = (gm < M && gk < K)
                              ? A[(long long)gm * lda + gk] : (unsigned short)0;
                }
            }
            if (fastB && fk) {
                if (b_nk) {
#pragma unroll
                    for (int it = 0; it < 2; ++it) {
                        int c = tid + it * 256;
                        int r = c >> 2, seg = (c & 3) * 8;
                        *(uint4*)&lB[r][seg] =
                            *(const uint4*)(Bm + (long long)(n0 + r) * ldb + k0 + seg);
                    }
                } else {
#pragma unroll
                    for (int it = 0; it < 4; ++it) {
                        int c = tid + it * 256;
                        int kk = c >> 5, nseg = (c & 31) * 4;
                        uint2 d = *(const uint2*)(Bm + (long long)(k0 + kk) * ldb + n0 + nseg);
                        lB[nseg + 0][kk] = (unsigned short)(d.x & 0xffffu);
                        lB[nseg + 1][kk] = (unsigned short)(d.x >> 16);
                        lB[nseg + 2][kk] = (unsigned short)(d.y & 0xffffu);
                        lB[nseg + 3][kk] = (unsigned short)(d.y >> 16);
                    }
                }
            } else {
                for (int i = tid; i < GBN * GBK; i += 256) {
                    int r  = i >> 5, kk = i & 31;
                    int gn = n0 + r, gk = k0 + kk;
                    unsigned short vs = 0;
                    if (gn < N && gk < K)
                        vs = b_nk ? Bm[(long long)gn * ldb + gk]
                                  : Bm[(long long)gk * ldb + gn];
                    lB[r][kk] = vs;
                }
            }
            __syncthreads();
            compute();
        }
    }

    // C/D layout: VGPR r, lane L -> (M = r + 8*(L>>4), N = L&15)
    int mh = (lane >> 4) * 8;
#pragma unroll
    for (int mt = 0; mt < 2; ++mt)
#pragma unroll
        for (int nt = 0; nt < 4; ++nt) {
            int col = n0 + wn + nt * 16 + lr;
            if (col >= N) continue;
            float bv = bias ? bias[col] : 0.f;
#pragma unroll
            for (int r = 0; r < 8; ++r) {
                int row = m0 + wm + mt * 16 + mh + r;
                if (row < M)
                    C[(long long)row * ldc + col] = acc[mt][nt][r] + bv;
            }
        }
}

// ---------------------------------------------------------------------------
// Elementwise / helper kernels
// ---------------------------------------------------------------------------
__global__ void cvt_bf16_kernel(const float* __restrict__ src,
                                unsigned short* __restrict__ dst, long long n)
{
    long long i = (long long)blockIdx.x * 256 + threadIdx.x;
    if (i < n) dst[i] = f2bf(src[i]);
}

// w3 [8,2730,1024] f32 -> [8,2752,1024] bf16 with zero-padded K rows
__global__ void cvt_w3_pad_kernel(const float* __restrict__ src,
                                  unsigned short* __restrict__ dst)
{
    int i = blockIdx.x * 256 + threadIdx.x;       // 8*2752*1024
    int c = i & 1023;
    int r = (i >> 10) % 2752;
    int l = i / (2752 * 1024);
    dst[i] = (r < 2730)
           ? f2bf(src[((long long)l * 2730 + r) * 1024 + c])
           : (unsigned short)0;
}

// x [B,256,256,3] f32 -> xp [B*256, 768] bf16  (patchify)
__global__ void patchify_kernel(const float* __restrict__ x,
                                unsigned short* __restrict__ xp)
{
    int i = blockIdx.x * 256 + threadIdx.x;       // 4096*768
    int col = i % 768; int tok = i / 768;
    int n = tok & 255; int b = tok >> 8;
    int py = col / 48; int rem = col - py * 48;
    int px = rem / 3;  int c  = rem - px * 3;
    int gy = n >> 4, gx = n & 15;
    long long src = (((long long)b * 256 + (gy * 16 + py)) * 256 + (gx * 16 + px)) * 3 + c;
    xp[i] = f2bf(x[src]);
}

__global__ void add_pos_kernel(float* __restrict__ H, const float* __restrict__ pos)
{
    int i = blockIdx.x * 256 + threadIdx.x;       // 4096*1024
    H[i] += pos[i & 0x3FFFF];                     // pos broadcast over batch
}

// timestep embedding: [16,256] bf16 = [cos(t*f) | sin(t*f)]
__global__ void time_embed_kernel(const float* __restrict__ t,
                                  unsigned short* __restrict__ e)
{
    int b = blockIdx.x; int j = threadIdx.x;      // 16 x 256
    float tv = t[b];
    int i = (j < 128) ? j : (j - 128);
    float freq = __expf(-9.210340372f * (float)i / 128.f);
    float a = tv * freq;
    e[b * 256 + j] = f2bf((j < 128) ? __cosf(a) : __sinf(a));
}

__global__ void silu_cvt_kernel(const float* __restrict__ x,
                                unsigned short* __restrict__ y, int n)
{
    int i = blockIdx.x * 256 + threadIdx.x;
    if (i < n) { float v = x[i]; y[i] = f2bf(v / (1.f + __expf(-v))); }
}

// fused RMSNorm (eps 1e-6) + adaLN modulate, f32 in -> bf16 out
__global__ __launch_bounds__(256) void rmsnorm_mod_kernel(
    const float* __restrict__ X, const float* __restrict__ w,
    const float* __restrict__ ada, int ada_ld, int shift_off, int scale_off,
    unsigned short* __restrict__ Out)
{
    __shared__ float red[256];
    int tok = blockIdx.x;                         // 0..4095
    int b = tok >> 8;
    const float* xr = X + (long long)tok * 1024;
    float v[4]; float s = 0.f;
#pragma unroll
    for (int j = 0; j < 4; ++j) { v[j] = xr[threadIdx.x + j * 256]; s += v[j] * v[j]; }
    red[threadIdx.x] = s;
    __syncthreads();
    for (int off = 128; off > 0; off >>= 1) {
        if (threadIdx.x < off) red[threadIdx.x] += red[threadIdx.x + off];
        __syncthreads();
    }
    float rms = rsqrtf(red[0] / 1024.f + 1e-6f);
    const float* ad = ada + (long long)b * ada_ld;
    unsigned short* orow = Out + (long long)tok * 1024;
#pragma unroll
    for (int j = 0; j < 4; ++j) {
        int d = threadIdx.x + j * 256;
        float y = v[j] * rms * w[d];
        y = y * (1.f + ad[scale_off + d]) + ad[shift_off + d];
        orow[d] = f2bf(y);
    }
}

// per-head q/k RMSNorm + RoPE, and v passthrough; qkv f32 [B,N,3,16,64]
// writes q,k bf16 in [B*H, N, 64]; v bf16 in [B*H, N, 128] (cols 64..127 = 0)
__global__ __launch_bounds__(64) void qk_rope_prep_kernel(
    const float* __restrict__ qkv, const float* __restrict__ qnw,
    const float* __restrict__ knw,
    unsigned short* __restrict__ Q, unsigned short* __restrict__ Kk,
    unsigned short* __restrict__ V)
{
    __shared__ float sq[64], sk[64], qh[64], kh[64];
    int idx = blockIdx.x;                         // b*4096 + n*16 + h
    int h = idx & 15;
    int n = (idx >> 4) & 255;
    int b = idx >> 12;
    int d = threadIdx.x;
    long long base = ((long long)(b * 256 + n)) * 3072 + h * 64;
    float q  = qkv[base + d];
    float k  = qkv[base + 1024 + d];
    float vv = qkv[base + 2048 + d];
    sq[d] = q * q; sk[d] = k * k;
    __syncthreads();
    for (int off = 32; off > 0; off >>= 1) {
        if (d < off) { sq[d] += sq[d + off]; sk[d] += sk[d + off]; }
        __syncthreads();
    }
    float qn = q * rsqrtf(sq[0] / 64.f + 1e-6f) * qnw[d];
    float kn = k * rsqrtf(sk[0] / 64.f + 1e-6f) * knw[d];
    qh[d] = qn; kh[d] = kn;
    __syncthreads();
    int fi = d & 15;
    float freq = __expf(-9.210340372f * (float)fi / 16.f); // 10000^(-fi/16)
    float p = (d < 32) ? (float)(n >> 4) : (float)(n & 15);
    float ang = p * freq;
    float cs = __cosf(ang), sn = __sinf(ang);
    float qr = (d < 32) ? -qh[d + 32] : qh[d - 32];
    float kr = (d < 32) ? -kh[d + 32] : kh[d - 32];
    long long ob  = ((long long)((b * 16 + h) * 256 + n)) * 64 + d;
    long long obv = ((long long)((b * 16 + h) * 256 + n)) * 128 + d;
    Q[ob]  = f2bf(qn * cs + qr * sn);
    Kk[ob] = f2bf(kn * cs + kr * sn);
    V[obv]      = f2bf(vv);
    V[obv + 64] = 0;                              // zero pad cols 64..127
}

// softmax over rows of 256 (1 wave = 1 row, 8 rows per block), bf16 out
__global__ __launch_bounds__(256) void softmax_kernel(
    const float* __restrict__ S, unsigned short* __restrict__ P, float scale)
{
    int warp = threadIdx.x >> 5, lane = threadIdx.x & 31;
    long long row = (long long)blockIdx.x * 8 + warp;
    const float* sr = S + row * 256;
    unsigned short* pr = P + row * 256;
    float e[8]; float m = -3.4e38f;
#pragma unroll
    for (int j = 0; j < 8; ++j) { e[j] = sr[lane + j * 32] * scale; m = fmaxf(m, e[j]); }
    for (int off = 16; off; off >>= 1) m = fmaxf(m, __shfl_xor(m, off, 32));
    float s = 0.f;
#pragma unroll
    for (int j = 0; j < 8; ++j) { e[j] = __expf(e[j] - m); s += e[j]; }
    for (int off = 16; off; off >>= 1) s += __shfl_xor(s, off, 32);
    float inv = 1.f / s;
#pragma unroll
    for (int j = 0; j < 8; ++j) pr[lane + j * 32] = f2bf(e[j] * inv);
}

// [B*H, N, 128] f32 (padded) -> [B, N, 1024] bf16
__global__ void merge_heads_kernel(const float* __restrict__ src,
                                   unsigned short* __restrict__ dst)
{
    int i = blockIdx.x * 256 + threadIdx.x;       // 4096*1024
    int d = i & 1023; int tok = i >> 10; int b = tok >> 8; int n = tok & 255;
    int h = d >> 6;  int hd = d & 63;
    dst[i] = f2bf(src[(((long long)(b * 16 + h)) * 256 + n) * 128 + hd]);
}

// H += gate(ada) * src
__global__ void gated_residual_kernel(float* __restrict__ H,
                                      const float* __restrict__ src,
                                      const float* __restrict__ ada,
                                      int ada_ld, int off)
{
    int i = blockIdx.x * 256 + threadIdx.x;       // 4096*1024
    int b = i >> 18; int d = i & 1023;
    H[i] += ada[(long long)b * ada_ld + off + d] * src[i];
}

// SwiGLU: out[row, j<2730] = silu(x1)*x2, K-padded to 2752 with zeros
__global__ void swiglu_kernel(const float* __restrict__ xw,
                              unsigned short* __restrict__ out)
{
    int i = blockIdx.x * 256 + threadIdx.x;       // 4096*2752
    int j = i % 2752; int row = i / 2752;
    unsigned short r = 0;
    if (j < 2730) {
        float x1 = xw[(long long)row * 5460 + j];
        float x2 = xw[(long long)row * 5460 + 2730 + j];
        r = f2bf((x1 / (1.f + __expf(-x1))) * x2);
    }
    out[i] = r;
}

// o [B,256,768] f32 -> out [B,256,256,3] f32 (unpatchify)
__global__ void unpatchify_kernel(const float* __restrict__ o,
                                  float* __restrict__ out)
{
    int i = blockIdx.x * 256 + threadIdx.x;       // 16*256*256*3
    int c  = i % 3;
    int w  = (i / 3) & 255;
    int hh = (i / 768) & 255;
    int b  = i / 196608;
    int gy = hh >> 4, py = hh & 15, gx = w >> 4, px = w & 15;
    long long src = ((long long)b * 256 + (gy * 16 + gx)) * 768 + py * 48 + px * 3 + c;
    out[i] = o[src];
}

// ---------------------------------------------------------------------------
// Host orchestration
// ---------------------------------------------------------------------------
extern "C" void kernel_launch(void* const* d_in, const int* in_sizes, int n_in,
                              void* d_out, int out_size, void* d_ws, size_t ws_size,
                              hipStream_t stream)
{
    (void)in_sizes; (void)n_in; (void)out_size; (void)ws_size;

    const float* x          = (const float*)d_in[0];
    const float* t          = (const float*)d_in[1];
    const float* pos        = (const float*)d_in[2];
    const float* compress_w = (const float*)d_in[3];
    const float* expand_w   = (const float*)d_in[4];
    const float* expand_b   = (const float*)d_in[5];
    const float* t_w1       = (const float*)d_in[6];
    const float* t_b1       = (const float*)d_in[7];
    const float* t_w2       = (const float*)d_in[8];
    const float* t_b2       = (const float*)d_in[9];
    const float* norm1_w    = (const float*)d_in[10];
    const float* norm2_w    = (const float*)d_in[11];
    const float* qn_w       = (const float*)d_in[12];
    const float* kn_w       = (const float*)d_in[13];
    const float* qkv_w      = (const float*)d_in[14];
    const float* qkv_b      = (const float*)d_in[15];
    const float* proj_w     = (const float*)d_in[16];
    const float* proj_b     = (const float*)d_in[17];
    const float* w12_w      = (const float*)d_in[18];
    const float* w12_b      = (const float*)d_in[19];
    const float* w3_w       = (const float*)d_in[20];
    const float* w3_b       = (const float*)d_in[21];
    const float* ada_w      = (const float*)d_in[22];
    const float* ada_b      = (const float*)d_in[23];
    const float* fnorm_w    = (const float*)d_in[24];
    const float* flin_w     = (const float*)d_in[25];
    const float* flin_b     = (const float*)d_in[26];
    const float* fada_w     = (const float*)d_in[27];
    const float* fada_b     = (const float*)d_in[28];

    // ---- workspace bump allocator -----------------------------------------
    char* wp = (char*)d_ws;
    auto alloc = [&](size_t bytes) -> void* {
        void* r = (void*)wp;
        wp += (bytes + 255) & ~(size_t)255;
        return r;
    };
    typedef unsigned short u16;

    // bf16 weights (stacked over DEPTH where applicable)
    u16* wb_compress = (u16*)alloc((size_t)768 * 128 * 2);
    u16* wb_expand   = (u16*)alloc((size_t)128 * 1024 * 2);
    u16* wb_tw1      = (u16*)alloc((size_t)256 * 1024 * 2);
    u16* wb_tw2      = (u16*)alloc((size_t)1024 * 1024 * 2);
    u16* wb_qkv      = (u16*)alloc((size_t)8 * 1024 * 3072 * 2);
    u16* wb_proj     = (u16*)alloc((size_t)8 * 1024 * 1024 * 2);
    u16* wb_w12      = (u16*)alloc((size_t)8 * 1024 * 5460 * 2);
    u16* wb_w3       = (u16*)alloc((size_t)8 * 2752 * 1024 * 2); // K-padded
    u16* wb_ada      = (u16*)alloc((size_t)8 * 1024 * 6144 * 2);
    u16* wb_flin     = (u16*)alloc((size_t)1024 * 768 * 2);
    u16* wb_fada     = (u16*)alloc((size_t)1024 * 2048 * 2);

    // activations
    u16*   xp_bf    = (u16*)alloc((size_t)4096 * 768 * 2);
    float* comp_f   = (float*)alloc((size_t)4096 * 128 * 4);
    u16*   compb_bf = (u16*)alloc((size_t)4096 * 128 * 2);
    float* h        = (float*)alloc((size_t)4096 * 1024 * 4);
    u16*   temb_bf  = (u16*)alloc((size_t)16 * 256 * 2);
    float* te1_f    = (float*)alloc((size_t)16 * 1024 * 4);
    u16*   te1_bf   = (u16*)alloc((size_t)16 * 1024 * 2);
    float* c_f      = (float*)alloc((size_t)16 * 1024 * 4);
    u16*   c_bf     = (u16*)alloc((size_t)16 * 1024 * 2);
    float* ada_f    = (float*)alloc((size_t)16 * 6144 * 4);
    u16*   act_bf   = (u16*)alloc((size_t)4096 * 1024 * 2);
    float* big_f    = (float*)alloc((size_t)4096 * 5460 * 4); // qkv / scores / w12
    u16*   q_bf     = (u16*)alloc((size_t)256 * 256 * 64 * 2);
    u16*   k_bf     = (u16*)alloc((size_t)256 * 256 * 64 * 2);
    u16*   v_bf     = (u16*)alloc((size_t)256 * 256 * 128 * 2);  // N-padded to 128
    u16*   probs_bf = (u16*)alloc((size_t)256 * 256 * 256 * 2);
    float* atno_f   = (float*)alloc((size_t)256 * 256 * 128 * 4); // N-padded to 128
    float* gout_f   = (float*)alloc((size_t)4096 * 1024 * 4);
    u16*   swig_bf  = (u16*)alloc((size_t)4096 * 2752 * 2);
    float* fada_f   = (float*)alloc((size_t)16 * 2048 * 4);
    float* o_f      = (float*)alloc((size_t)4096 * 768 * 4);

    auto cvt = [&](const float* src, u16* dst, long long n) {
        int blocks = (int)((n + 255) / 256);
        cvt_bf16_kernel<<<blocks, 256, 0, stream>>>(src, dst, n);
    };
    auto gemm = [&](const u16* A, const u16* Bp, const float* bias, float* C,
                    int M, int N, int K, int lda, int ldb, int ldc,
                    long long aB, long long bB, long long cB, int batches, int bnk) {
        dim3 g((N + GBN - 1) / GBN, (M + GBM - 1) / GBM, batches);
        gemm_bf16_kernel<<<g, 256, 0, stream>>>(A, Bp, bias, C, M, N, K,
                                                lda, ldb, ldc, aB, bB, cB, bnk);
    };

    // ---- convert weights to bf16 ------------------------------------------
    cvt(compress_w, wb_compress, 768 * 128);
    cvt(expand_w,   wb_expand,   128 * 1024);
    cvt(t_w1,       wb_tw1,      256 * 1024);
    cvt(t_w2,       wb_tw2,      1024 * 1024);
    cvt(qkv_w,      wb_qkv,      (long long)8 * 1024 * 3072);
    cvt(proj_w,     wb_proj,     (long long)8 * 1024 * 1024);
    cvt(w12_w,      wb_w12,      (long long)8 * 1024 * 5460);
    cvt_w3_pad_kernel<<<88064, 256, 0, stream>>>(w3_w, wb_w3);
    cvt(ada_w,      wb_ada,      (long long)8 * 1024 * 6144);
    cvt(flin_w,     wb_flin,     1024 * 768);
    cvt(fada_w,     wb_fada,     1024 * 2048);

    // ---- patch embed: xp @ compress @ expand + pos ------------------------
    patchify_kernel<<<12288, 256, 0, stream>>>(x, xp_bf);
    gemm(xp_bf, wb_compress, nullptr, comp_f, 4096, 128, 768, 768, 128, 128, 0, 0, 0, 1, 0);
    cvt(comp_f, compb_bf, 4096 * 128);
    gemm(compb_bf, wb_expand, expand_b, h, 4096, 1024, 128, 128, 1024, 1024, 0, 0, 0, 1, 0);
    add_pos_kernel<<<16384, 256, 0, stream>>>(h, pos);

    // ---- timestep embedding -> c ------------------------------------------
    time_embed_kernel<<<16, 256, 0, stream>>>(t, temb_bf);
    gemm(temb_bf, wb_tw1, t_b1, te1_f, 16, 1024, 256, 256, 1024, 1024, 0, 0, 0, 1, 0);
    silu_cvt_kernel<<<64, 256, 0, stream>>>(te1_f, te1_bf, 16 * 1024);
    gemm(te1_bf, wb_tw2, t_b2, c_f, 16, 1024, 1024, 1024, 1024, 1024, 0, 0, 0, 1, 0);
    cvt(c_f, c_bf, 16 * 1024);

    // ---- transformer blocks ------------------------------------------------
    for (int i = 0; i < 8; ++i) {
        // adaLN: [16,6144]
        gemm(c_bf, wb_ada + (size_t)i * 1024 * 6144, ada_b + (size_t)i * 6144,
             ada_f, 16, 6144, 1024, 1024, 6144, 6144, 0, 0, 0, 1, 0);

        // attention branch
        rmsnorm_mod_kernel<<<4096, 256, 0, stream>>>(
            h, norm1_w + (size_t)i * 1024, ada_f, 6144, 0, 1024, act_bf);
        gemm(act_bf, wb_qkv + (size_t)i * 1024 * 3072, qkv_b + (size_t)i * 3072,
             big_f, 4096, 3072, 1024, 1024, 3072, 3072, 0, 0, 0, 1, 0);
        qk_rope_prep_kernel<<<65536, 64, 0, stream>>>(
            big_f, qn_w + (size_t)i * 64, kn_w + (size_t)i * 64, q_bf, k_bf, v_bf);
        // scores = q @ k^T  (k stored [N,64] -> b_nk=1), batched over B*H=256
        gemm(q_bf, k_bf, nullptr, big_f, 256, 256, 64, 64, 64, 256,
             256 * 64, 256 * 64, 65536LL, 256, 1);
        softmax_kernel<<<8192, 256, 0, stream>>>(big_f, probs_bf, 0.125f);
        // out = probs @ v  (v padded to 128 cols so staging stays vectorized)
        gemm(probs_bf, v_bf, nullptr, atno_f, 256, 128, 256, 256, 128, 128,
             65536LL, 256 * 128, 256 * 128, 256, 0);
        merge_heads_kernel<<<16384, 256, 0, stream>>>(atno_f, act_bf);
        gemm(act_bf, wb_proj + (size_t)i * 1024 * 1024, proj_b + (size_t)i * 1024,
             gout_f, 4096, 1024, 1024, 1024, 1024, 1024, 0, 0, 0, 1, 0);
        gated_residual_kernel<<<16384, 256, 0, stream>>>(h, gout_f, ada_f, 6144, 2048);

        // MLP (SwiGLU) branch
        rmsnorm_mod_kernel<<<4096, 256, 0, stream>>>(
            h, norm2_w + (size_t)i * 1024, ada_f, 6144, 3072, 4096, act_bf);
        gemm(act_bf, wb_w12 + (size_t)i * 1024 * 5460, w12_b + (size_t)i * 5460,
             big_f, 4096, 5460, 1024, 1024, 5460, 5460, 0, 0, 0, 1, 0);
        swiglu_kernel<<<44032, 256, 0, stream>>>(big_f, swig_bf);
        gemm(swig_bf, wb_w3 + (size_t)i * 2752 * 1024, w3_b + (size_t)i * 1024,
             gout_f, 4096, 1024, 2752, 2752, 1024, 1024, 0, 0, 0, 1, 0);
        gated_residual_kernel<<<16384, 256, 0, stream>>>(h, gout_f, ada_f, 6144, 5120);
    }

    // ---- final layer -------------------------------------------------------
    gemm(c_bf, wb_fada, fada_b, fada_f, 16, 2048, 1024, 1024, 2048, 2048, 0, 0, 0, 1, 0);
    rmsnorm_mod_kernel<<<4096, 256, 0, stream>>>(h, fnorm_w, fada_f, 2048, 0, 1024, act_bf);
    gemm(act_bf, wb_flin, flin_b, o_f, 4096, 768, 1024, 1024, 768, 768, 0, 0, 0, 1, 0);
    unpatchify_kernel<<<12288, 256, 0, stream>>>(o_f, (float*)d_out);
}
